// ViT_49649821942415
// MI455X (gfx1250) — matrix operand order repaired and checked by reading, not compiled, not run on hardware
//
#include <hip/hip_runtime.h>
#include <hip/hip_bf16.h>

// ---------------- constants ----------------
#define ZB   128
#define ZHW  224
#define ZP   16
#define ZD   512
#define ZNH  8
#define ZL   6
#define ZM   2048
#define ZNC  10
#define ZGP  14
#define ZT   197            // ZGP*ZGP + 1
#define ZTP  224            // T padded to multiple of 32 (tile-friendly)
#define ZNP  196            // patches per image
#define ZPK  256            // P*P
#define ZHTP (ZNH*ZTP)      // 1792 padded head-concat length

typedef __attribute__((ext_vector_type(16))) __bf16 v16bf;
typedef __attribute__((ext_vector_type(8)))  __bf16 v8bf;
typedef __attribute__((ext_vector_type(8)))  float  v8f;

// ---------------- generic batched bf16 WMMA GEMM ----------------
// Requirements (guaranteed by caller): K % 32 == 0, lda % 8 == 0, ldb % 8 == 0,
// base pointers 16B-aligned, OOB M/N-edge *reads* land in workspace slack.
// C[z] = act( alpha * A[z](MxK) * B[z](KxN) + bias (+ resid[z]) )
// TRANSB==0 -> B[k*ldb+n], TRANSB==1 -> B[n*ldb+k]
// batch offsets: off = (z/bdiv)*s1 + (z%bdiv)*s2   (element strides)
struct GemmP {
  const __bf16* A;
  const __bf16* Bm;
  const float*  bias;   // nullable
  const float*  resid;  // nullable (indexed like C)
  float*        outF;   // nullable
  __bf16*       outB;   // nullable
  int M, N, K;
  int lda, ldb, ldc;
  int biasMode;         // 0 none, 1 per-n, 2 per-m
  int act;              // 0 none, 1 gelu(exact)
  float alpha;
  int bdiv;
  long long sA1, sA2, sB1, sB2, sC1, sC2;
};

#define BM 64
#define BN 128
#define BK 32
#define LDS_PAD 8                    // keeps rows 16B-aligned
#define LDW (BK + LDS_PAD)           // 40 elems per row
#define A_ELEMS (BM * LDW)           // 2560
#define B_ELEMS (BN * LDW)           // 5120
#define STAGE  (A_ELEMS + B_ELEMS)   // elems per double-buffer stage

#define ASYNC_B128(ldsoff, gaddr) \
  asm volatile("global_load_async_to_lds_b128 %0, %1, off" :: "v"(ldsoff), "v"(gaddr) : "memory")

template <int TRANSB>
__global__ __launch_bounds__(128) void k_gemm_wmma(GemmP p) {
  __shared__ __attribute__((aligned(16))) __bf16 smem[2 * STAGE];

  const int z = blockIdx.z;
  const long long offA = (long long)(z / p.bdiv) * p.sA1 + (long long)(z % p.bdiv) * p.sA2;
  const long long offB = (long long)(z / p.bdiv) * p.sB1 + (long long)(z % p.bdiv) * p.sB2;
  const long long offC = (long long)(z / p.bdiv) * p.sC1 + (long long)(z % p.bdiv) * p.sC2;
  const __bf16* A = p.A  + offA;
  const __bf16* B = p.Bm + offB;
  const int lda = p.lda, ldb = p.ldb, K = p.K;

  const int m0 = blockIdx.y * BM;
  const int n0 = blockIdx.x * BN;
  const int tid  = threadIdx.x;
  const int lane = tid & 31;
  const int wave = tid >> 5;
  const int wm = (wave >> 1) * 32;   // wave row offset in tile
  const int wn = (wave & 1) * 64;    // wave col offset in tile
  const int lrow = lane & 15;
  const int lhiA = (lane & 16) ? 8 : 0;    // A/C hi-half offset
  const int lhiB = (lane & 16) ? 16 : 0;   // unused (B reads use lhiA)

  // staging coordinates (uniform control flow, no guards)
  const int sr  = tid >> 2;          // 0..31
  const int sc8 = (tid & 3) * 8;     // 0,8,16,24

  // stage tile t into buffer dbuf (A via async DMA; B async if TRANSB else scatter)
  auto stage = [&](int t, int dbuf) {
    const int k0 = t * BK;
    const unsigned sbase = (unsigned)dbuf * STAGE;
    #pragma unroll
    for (int rep = 0; rep < 2; rep++) {
      const int row = sr + rep * 32;
      const unsigned lo = (sbase + (unsigned)row * LDW + sc8) * 2u;
      const void* ga = (const void*)&A[(long long)(m0 + row) * lda + (k0 + sc8)];
      ASYNC_B128(lo, ga);
    }
    if (TRANSB) {
      #pragma unroll
      for (int rep = 0; rep < 4; rep++) {
        const int n = sr + rep * 32;
        const unsigned lo = (sbase + A_ELEMS + (unsigned)n * LDW + sc8) * 2u;
        const void* ga = (const void*)&B[(long long)(n0 + n) * ldb + (k0 + sc8)];
        ASYNC_B128(lo, ga);
      }
    } else {
      #pragma unroll
      for (int rep = 0; rep < 4; rep++) {
        const int n8 = (tid & 3) * 8 + rep * 32;
        const v8bf vv = *(const v8bf*)&B[(long long)(k0 + sr) * ldb + (n0 + n8)];
        #pragma unroll
        for (int e = 0; e < 8; e++) smem[sbase + A_ELEMS + (unsigned)(n8 + e) * LDW + sr] = vv[e];
      }
    }
  };

  v8f acc[2][4] = {};
  const int nt = K >> 5;      // K/32 tiles
  stage(0, 0);                // prologue

  for (int t = 0; t < nt; ++t) {
    const int cur = t & 1;
    // this tile's async loads (issued last iteration / prologue) must land
    asm volatile("s_wait_asynccnt 0x0" ::: "memory");
    __syncthreads();          // all waves' A/B staging visible

    const __bf16* Ab = &smem[(unsigned)cur * STAGE];
    const __bf16* Bb = Ab + A_ELEMS;
    v16bf afr[2], bfr[4];
    #pragma unroll
    for (int fi = 0; fi < 2; fi++) {
      const int mrow = wm + fi * 16 + lrow;
      const v8bf a_lo = *(const v8bf*)&Ab[(unsigned)mrow * LDW + lhiA];       // K=lhiA..+7
      const v8bf a_hi = *(const v8bf*)&Ab[(unsigned)mrow * LDW + lhiA + 16];  // K=16+lhiA..
      afr[fi] = __builtin_shufflevector(a_lo, a_hi, 0,1,2,3,4,5,6,7,8,9,10,11,12,13,14,15);
    }
    #pragma unroll
    for (int fj = 0; fj < 4; fj++) {
      const int ncol = wn + fj * 16 + lrow;
      const v8bf b_lo = *(const v8bf*)&Bb[(unsigned)ncol * LDW + lhiA];       // K=lhiA..+7
      const v8bf b_hi = *(const v8bf*)&Bb[(unsigned)ncol * LDW + lhiA + 16];  // K=16+lhiA..
      bfr[fj] = __builtin_shufflevector(b_lo, b_hi, 0,1,2,3,4,5,6,7,8,9,10,11,12,13,14,15);
    }
    __syncthreads();          // everyone consumed buf[cur^1]-adjacent hazards; safe to refill

    if (t + 1 < nt) stage(t + 1, cur ^ 1);   // DMA next tile under the WMMAs

    #pragma unroll
    for (int fi = 0; fi < 2; fi++)
      #pragma unroll
      for (int fj = 0; fj < 4; fj++)
        acc[fi][fj] = __builtin_amdgcn_wmma_f32_16x16x32_bf16(
            false, afr[fi], false, bfr[fj], (short)0, acc[fi][fj], false, false);
  }

  // epilogue: C VGPR i -> M = i (+8 hi half), N = lane&15
  #pragma unroll
  for (int fi = 0; fi < 2; fi++) {
    #pragma unroll
    for (int fj = 0; fj < 4; fj++) {
      #pragma unroll
      for (int i = 0; i < 8; i++) {
        const int m = m0 + wm + fi * 16 + i + lhiA;
        const int n = n0 + wn + fj * 16 + lrow;
        if (m < p.M && n < p.N) {
          float v = acc[fi][fj][i] * p.alpha;
          if (p.biasMode == 1)      v += p.bias[n];
          else if (p.biasMode == 2) v += p.bias[m];
          const long long ci = offC + (long long)m * p.ldc + n;
          if (p.resid) v += p.resid[ci];
          if (p.act == 1) v = 0.5f * v * (1.0f + erff(v * 0.70710678118654752f));
          if (p.outF) p.outF[ci] = v;
          if (p.outB) p.outB[ci] = (__bf16)v;
        }
      }
    }
  }
}

// ---------------- layernorm (row-per-block) ----------------
__global__ __launch_bounds__(256) void k_layernorm(const float* __restrict__ x,
                                                   const float* __restrict__ g,
                                                   const float* __restrict__ b,
                                                   float* outF, __bf16* outB, int ncols) {
  const long long row = blockIdx.x;
  const float* xr = x + row * ncols;
  __shared__ float red[256];
  const int tid = threadIdx.x;

  float s = 0.0f;
  for (int i = tid; i < ncols; i += 256) s += xr[i];
  red[tid] = s; __syncthreads();
  for (int st = 128; st > 0; st >>= 1) { if (tid < st) red[tid] += red[tid + st]; __syncthreads(); }
  const float mean = red[0] / ncols;
  __syncthreads();

  s = 0.0f;
  for (int i = tid; i < ncols; i += 256) { float d = xr[i] - mean; s += d * d; }
  red[tid] = s; __syncthreads();
  for (int st = 128; st > 0; st >>= 1) { if (tid < st) red[tid] += red[tid + st]; __syncthreads(); }
  const float inv = rsqrtf(red[0] / ncols + 1e-5f);
  __syncthreads();

  for (int i = tid; i < ncols; i += 256) {
    float v = (xr[i] - mean) * inv * g[i] + b[i];
    if (outF) outF[row * ncols + i] = v;
    outB[row * ncols + i] = (__bf16)v;
  }
}

// ---------- softmax over padded score rows: grid = B*NH*ZTP ----------
__global__ __launch_bounds__(128) void k_softmax(const float* __restrict__ in,
                                                 __bf16* __restrict__ out) {
  const long long z = blockIdx.x;
  const long long bh = z / ZTP;
  const int t = (int)(z % ZTP);
  const long long rowoff = (bh * ZTP + t) * (long long)ZTP;
  __bf16* o = out + rowoff;
  const int tid = threadIdx.x;
  if (t >= ZT) {
    for (int i = tid; i < ZTP; i += 128) o[i] = (__bf16)0.0f;
    return;
  }
  const float* r = in + rowoff;
  __shared__ float red[128];

  float mx = -1e30f;
  for (int i = tid; i < ZT; i += 128) mx = fmaxf(mx, r[i]);
  red[tid] = mx; __syncthreads();
  for (int st = 64; st > 0; st >>= 1) { if (tid < st) red[tid] = fmaxf(red[tid], red[tid + st]); __syncthreads(); }
  mx = red[0]; __syncthreads();

  float s = 0.0f;
  for (int i = tid; i < ZT; i += 128) s += __expf(r[i] - mx);
  red[tid] = s; __syncthreads();
  for (int st = 64; st > 0; st >>= 1) { if (tid < st) red[tid] += red[tid + st]; __syncthreads(); }
  const float inv = 1.0f / red[0];
  __syncthreads();

  for (int i = tid; i < ZTP; i += 128)
    o[i] = (i < ZT) ? (__bf16)(__expf(r[i] - mx) * inv) : (__bf16)0.0f;
}

// ---------------- weight preprocessing ----------------
__global__ void k_transpose_bf(const float* __restrict__ in, __bf16* __restrict__ out,
                               int R, int C, long long total) {
  long long idx = (long long)blockIdx.x * 256 + threadIdx.x;
  if (idx >= total) return;
  long long rc = (long long)R * C;
  long long bz = idx / rc;
  long long rem = idx - bz * rc;
  int r = (int)(rem / C), c = (int)(rem % C);
  out[bz * rc + (long long)c * R + r] = (__bf16)in[idx];
}

// qkv: in (L,NH,D(e),D(d)) -> out (L, D(d), NH*D(e))
__global__ void k_qkv_t(const float* __restrict__ in, __bf16* __restrict__ out, long long total) {
  long long idx = (long long)blockIdx.x * 256 + threadIdx.x;
  if (idx >= total) return;
  int d = (int)(idx % ZD);
  long long t1 = idx / ZD;
  int e = (int)(t1 % ZD);
  long long t2 = t1 / ZD;
  int h = (int)(t2 % ZNH);
  int l = (int)(t2 / ZNH);
  out[(((long long)l * ZD + d) * ZNH + h) * ZD + e] = (__bf16)in[idx];
}

// map_w (L, T, NH*T) -> padded bf16 (L, ZTP rows, NH*ZTP), zeros in pads
__global__ void k_mapw_pad(const float* __restrict__ in, __bf16* __restrict__ out, long long total) {
  long long idx = (long long)blockIdx.x * 256 + threadIdx.x;
  if (idx >= total) return;
  int col = (int)(idx % ZHTP);
  long long t1 = idx / ZHTP;
  int t = (int)(t1 % ZTP);
  int l = (int)(t1 / ZTP);
  int h = col / ZTP, s = col % ZTP;
  float v = 0.0f;
  if (t < ZT && s < ZT)
    v = in[((long long)l * ZT + t) * (ZNH * ZT) + h * ZT + s];
  out[idx] = (__bf16)v;
}

// patch extraction: images (B,1,224,224) -> (B,196,256) bf16
__global__ void k_patches(const float* __restrict__ img, __bf16* __restrict__ out, long long total) {
  long long idx = (long long)blockIdx.x * 256 + threadIdx.x;
  if (idx >= total) return;
  int i = (int)(idx % ZPK);
  long long t1 = idx / ZPK;
  int pch = (int)(t1 % ZNP);
  int b = (int)(t1 / ZNP);
  int py = i / ZP, px = i % ZP;
  int gy = pch / ZGP, gx = pch % ZGP;
  out[idx] = (__bf16)img[((long long)b * ZHW + gy * ZP + py) * ZHW + gx * ZP + px];
}

// class token: x[b, 0, :] = ct[:]
__global__ void k_clstok(const float* __restrict__ ct, float* __restrict__ x, long long total) {
  long long idx = (long long)blockIdx.x * 256 + threadIdx.x;
  if (idx >= total) return;
  int d = (int)(idx % ZD);
  long long b = idx / ZD;
  x[b * (long long)ZT * ZD + d] = ct[d];
}

// ---------------- classifier head + softmax ----------------
__global__ __launch_bounds__(256) void k_classifier(const float* __restrict__ x,
                                                    const float* __restrict__ w,
                                                    const float* __restrict__ bias,
                                                    float* __restrict__ out) {
  const int bi = blockIdx.x;
  const float* xr = x + (long long)bi * ZT * ZD;
  __shared__ float red[256];
  __shared__ float logits[ZNC];
  const int tid = threadIdx.x;
  for (int c = 0; c < ZNC; c++) {
    float s = 0.0f;
    for (int i = tid; i < ZD; i += 256) s += xr[i] * w[c * ZD + i];
    red[tid] = s; __syncthreads();
    for (int st = 128; st > 0; st >>= 1) { if (tid < st) red[tid] += red[tid + st]; __syncthreads(); }
    if (tid == 0) logits[c] = red[0] + bias[c];
    __syncthreads();
  }
  if (tid == 0) {
    float mx = -1e30f;
    for (int c = 0; c < ZNC; c++) mx = fmaxf(mx, logits[c]);
    float s = 0.0f;
    for (int c = 0; c < ZNC; c++) { float e = __expf(logits[c] - mx); logits[c] = e; s += e; }
    for (int c = 0; c < ZNC; c++) out[bi * ZNC + c] = logits[c] / s;
  }
}

// ---------------- host-side launch helpers ----------------
static void gemm(hipStream_t st, const __bf16* A, const __bf16* B, const float* bias,
                 const float* resid, float* outF, __bf16* outB,
                 int M, int N, int K, int lda, int ldb, int ldc,
                 int transB, int biasMode, int act, float alpha,
                 int batch, int bdiv,
                 long long sA1, long long sA2, long long sB1, long long sB2,
                 long long sC1, long long sC2) {
  GemmP p;
  p.A = A; p.Bm = B; p.bias = bias; p.resid = resid; p.outF = outF; p.outB = outB;
  p.M = M; p.N = N; p.K = K; p.lda = lda; p.ldb = ldb; p.ldc = ldc;
  p.biasMode = biasMode; p.act = act; p.alpha = alpha;
  p.bdiv = bdiv; p.sA1 = sA1; p.sA2 = sA2; p.sB1 = sB1; p.sB2 = sB2; p.sC1 = sC1; p.sC2 = sC2;
  dim3 grid((N + BN - 1) / BN, (M + BM - 1) / BM, batch);
  if (transB) k_gemm_wmma<1><<<grid, dim3(128), 0, st>>>(p);
  else        k_gemm_wmma<0><<<grid, dim3(128), 0, st>>>(p);
}

static inline long long ceil_div(long long a, long long b) { return (a + b - 1) / b; }

extern "C" void kernel_launch(void* const* d_in, const int* in_sizes, int n_in,
                              void* d_out, int out_size, void* d_ws, size_t ws_size,
                              hipStream_t stream) {
  (void)in_sizes; (void)n_in; (void)out_size; (void)ws_size;
  const float* images     = (const float*)d_in[0];
  const float* lin_map_w  = (const float*)d_in[1];
  const float* lin_map_b  = (const float*)d_in[2];
  const float* class_tok  = (const float*)d_in[3];
  const float* ln1_g      = (const float*)d_in[4];
  const float* ln1_b      = (const float*)d_in[5];
  const float* ln2_g      = (const float*)d_in[6];
  const float* ln2_b      = (const float*)d_in[7];
  const float* qw         = (const float*)d_in[8];
  const float* qb         = (const float*)d_in[9];
  const float* kw         = (const float*)d_in[10];
  const float* kb         = (const float*)d_in[11];
  const float* vw         = (const float*)d_in[12];
  const float* vb         = (const float*)d_in[13];
  const float* map_w      = (const float*)d_in[14];
  const float* map_b      = (const float*)d_in[15];
  const float* mlp_w1     = (const float*)d_in[16];
  const float* mlp_b1     = (const float*)d_in[17];
  const float* mlp_w2     = (const float*)d_in[18];
  const float* mlp_b2     = (const float*)d_in[19];
  const float* cls_w      = (const float*)d_in[20];
  const float* cls_b      = (const float*)d_in[21];
  float* out = (float*)d_out;

  // ---- workspace bump allocator ----
  char* base = (char*)d_ws;
  size_t off = 0;
  auto alloc = [&](size_t bytes) -> void* {
    void* p = base + off;
    off += (bytes + 255) & ~(size_t)255;
    return p;
  };
  const long long BT   = (long long)ZB * ZT;           // 25216 (=64*394)
  const long long NHD  = (long long)ZNH * ZD;          // 4096
  const long long QKVN = BT * NHD;
  const long long SCP  = (long long)ZB * ZNH * ZTP * ZTP;
  const long long RESP = (long long)ZB * ZNH * ZTP * ZD;

  __bf16* lin_w_t  = (__bf16*)alloc((size_t)ZPK * ZD * 2);
  __bf16* wq_t     = (__bf16*)alloc((size_t)ZL * ZD * NHD * 2);
  __bf16* wk_t     = (__bf16*)alloc((size_t)ZL * ZD * NHD * 2);
  __bf16* wv_t     = (__bf16*)alloc((size_t)ZL * ZD * NHD * 2);
  __bf16* mapw_p   = (__bf16*)alloc((size_t)ZL * ZTP * ZHTP * 2);
  __bf16* w1_t     = (__bf16*)alloc((size_t)ZL * ZD * ZM * 2);
  __bf16* w2_t     = (__bf16*)alloc((size_t)ZL * ZM * ZD * 2);
  __bf16* patches  = (__bf16*)alloc((size_t)ZB * ZNP * ZPK * 2);
  float*  x        = (float*) alloc((size_t)BT * ZD * 4);
  __bf16* h_bf     = (__bf16*)alloc((size_t)BT * ZD * 2);   // reused as y_bf
  float*  y_f      = (float*) alloc((size_t)BT * ZD * 4);
  __bf16* q_bf     = (__bf16*)alloc((size_t)QKVN * 2);
  __bf16* kk_bf    = (__bf16*)alloc((size_t)QKVN * 2);
  __bf16* v_bf     = (__bf16*)alloc((size_t)QKVN * 2);
  float*  scores   = (float*) alloc((size_t)SCP * 4);
  __bf16* attn_bf  = (__bf16*)alloc((size_t)SCP * 2);
  __bf16* res_bf   = (__bf16*)alloc((size_t)RESP * 2);
  __bf16* mid_bf   = (__bf16*)alloc((size_t)BT * ZM * 2);
  (void)alloc((size_t)8 << 20);   // slack for unguarded M/N-edge tile reads

  // ---- weight preprocessing ----
  {
    long long tot = (long long)ZD * ZPK;
    k_transpose_bf<<<dim3((unsigned)ceil_div(tot, 256)), 256, 0, stream>>>(lin_map_w, lin_w_t, ZD, ZPK, tot);
  }
  {
    long long tot = (long long)ZL * ZNH * ZD * ZD;
    dim3 g((unsigned)ceil_div(tot, 256));
    k_qkv_t<<<g, 256, 0, stream>>>(qw, wq_t, tot);
    k_qkv_t<<<g, 256, 0, stream>>>(kw, wk_t, tot);
    k_qkv_t<<<g, 256, 0, stream>>>(vw, wv_t, tot);
  }
  {
    long long tot = (long long)ZL * ZTP * ZHTP;
    k_mapw_pad<<<dim3((unsigned)ceil_div(tot, 256)), 256, 0, stream>>>(map_w, mapw_p, tot);
  }
  {
    long long tot = (long long)ZL * ZM * ZD;
    k_transpose_bf<<<dim3((unsigned)ceil_div(tot, 256)), 256, 0, stream>>>(mlp_w1, w1_t, ZM, ZD, tot);
    k_transpose_bf<<<dim3((unsigned)ceil_div(tot, 256)), 256, 0, stream>>>(mlp_w2, w2_t, ZD, ZM, tot);
  }

  // ---- patch extraction + class token + patch embedding ----
  {
    long long tot = (long long)ZB * ZNP * ZPK;
    k_patches<<<dim3((unsigned)ceil_div(tot, 256)), 256, 0, stream>>>(images, patches, tot);
    long long tc = (long long)ZB * ZD;
    k_clstok<<<dim3((unsigned)ceil_div(tc, 256)), 256, 0, stream>>>(class_tok, x, tc);
  }
  gemm(stream, patches, lin_w_t, lin_map_b, nullptr, x + ZD, nullptr,
       ZNP, ZD, ZPK, ZPK, ZD, ZD, 0, 1, 0, 1.0f,
       ZB, 1,
       (long long)ZNP * ZPK, 0, 0, 0, (long long)ZT * ZD, 0);

  const float attn_scale = 1.0f / 14.0f;  // 1/sqrt(T-1)

  for (int l = 0; l < ZL; l++) {
    k_layernorm<<<dim3((unsigned)BT), 256, 0, stream>>>(x, ln1_g + (long long)l * ZD,
                                                        ln1_b + (long long)l * ZD,
                                                        nullptr, h_bf, ZD);
    // QKV: (BT x 512) x (512 x 4096) -> layout (B,T,NH,D)
    gemm(stream, h_bf, wq_t + (long long)l * ZD * NHD, qb + (long long)l * NHD, nullptr,
         nullptr, q_bf, (int)BT, (int)NHD, ZD, ZD, (int)NHD, (int)NHD, 0, 1, 0, 1.0f,
         1, 1, 0, 0, 0, 0, 0, 0);
    gemm(stream, h_bf, wk_t + (long long)l * ZD * NHD, kb + (long long)l * NHD, nullptr,
         nullptr, kk_bf, (int)BT, (int)NHD, ZD, ZD, (int)NHD, (int)NHD, 0, 1, 0, 1.0f,
         1, 1, 0, 0, 0, 0, 0, 0);
    gemm(stream, h_bf, wv_t + (long long)l * ZD * NHD, vb + (long long)l * NHD, nullptr,
         nullptr, v_bf, (int)BT, (int)NHD, ZD, ZD, (int)NHD, (int)NHD, 0, 1, 0, 1.0f,
         1, 1, 0, 0, 0, 0, 0, 0);

    // scores[z=(b,h)] (padded 224x224) = scale * q (224x512) * k^T   [async TRANSB=1 path]
    gemm(stream, q_bf, kk_bf, nullptr, nullptr, scores, nullptr,
         ZTP, ZTP, ZD, (int)NHD, (int)NHD, ZTP, 1, 0, 0, attn_scale,
         ZB * ZNH, ZNH,
         (long long)ZT * NHD, ZD,
         (long long)ZT * NHD, ZD,
         (long long)ZNH * ZTP * ZTP, (long long)ZTP * ZTP);

    k_softmax<<<dim3((unsigned)((long long)ZB * ZNH * ZTP)), 128, 0, stream>>>(scores, attn_bf);

    // av[z] = attn (224x224) * v (224x512) -> res (B,NH,224,512)
    gemm(stream, attn_bf, v_bf, nullptr, nullptr, nullptr, res_bf,
         ZTP, ZD, ZTP, ZTP, (int)NHD, ZD, 0, 0, 0, 1.0f,
         ZB * ZNH, ZNH,
         (long long)ZNH * ZTP * ZTP, (long long)ZTP * ZTP,
         (long long)ZT * NHD, ZD,
         (long long)ZNH * ZTP * ZD, (long long)ZTP * ZD);

    // msa: per-batch mapw_p (197x1792) * res (1792x512) + map_b[t] + x -> x
    gemm(stream, mapw_p + (long long)l * ZTP * ZHTP, res_bf, map_b + (long long)l * ZT, x,
         x, nullptr, ZT, ZD, ZHTP, ZHTP, ZD, ZD, 0, 2, 0, 1.0f,
         ZB, 1,
         0, 0, (long long)ZNH * ZTP * ZD, 0, (long long)ZT * ZD, 0);

    k_layernorm<<<dim3((unsigned)BT), 256, 0, stream>>>(x, ln2_g + (long long)l * ZD,
                                                        ln2_b + (long long)l * ZD,
                                                        y_f, h_bf, ZD);
    // MLP1: (BT x 512) x (512 x 2048) + b1, GELU -> mid
    gemm(stream, h_bf, w1_t + (long long)l * ZD * ZM, mlp_b1 + (long long)l * ZM, nullptr,
         nullptr, mid_bf, (int)BT, ZM, ZD, ZD, ZM, ZM, 0, 1, 1, 1.0f,
         1, 1, 0, 0, 0, 0, 0, 0);
    // MLP2: (BT x 2048) x (2048 x 512) + b2 + y -> x
    gemm(stream, mid_bf, w2_t + (long long)l * ZM * ZD, mlp_b2 + (long long)l * ZD, y_f,
         x, nullptr, (int)BT, ZD, ZM, ZM, ZD, ZD, 0, 1, 0, 1.0f,
         1, 1, 0, 0, 0, 0, 0, 0);
  }

  k_classifier<<<dim3(ZB), 256, 0, stream>>>(x, cls_w, cls_b, out);
}
